// Token_Choice_12223476924649
// MI455X (gfx1250) — compile-verified, matrix operand order checked by reading, not run on hardware
//
#include <hip/hip_runtime.h>
#include <hip/hip_bf16.h>
#include <math.h>

typedef __attribute__((ext_vector_type(16))) _Float16 v16h;
typedef __attribute__((ext_vector_type(8)))  float    v8f;

// ---------------------------------------------------------------------------
// Fragment loader for V_WMMA_F32_16X16X32_F16 (wave32), ISA 7.12.2 layout:
// lanes 0-15 hold row M=lane, elements 0..7 = K{0..7}, 8..15 = K{16..23};
// lanes 16-31 hold row M=lane-16 with K{8..15} / K{24..31}.
// Works for A tiles and (with a transposed [n][k] LDS image) for B tiles.
// ld must be a multiple of 8 halves so each b128 read is 16B aligned; strides
// 1032/136 shift rows by 4 banks -> conflict-free across the 16 lane rows.
// ---------------------------------------------------------------------------
__device__ inline v16h ldsA(const _Float16* s, int ld, int k0, int lane) {
  const int row = lane & 15;
  const int kh  = (lane >> 4) << 3;                 // 0 or 8
  const _Float16* p = s + row * ld + k0 + kh;
  v16h a;
#pragma unroll
  for (int i = 0; i < 8; ++i) a[i] = p[i];          // K = kh + 0..7
#pragma unroll
  for (int i = 0; i < 8; ++i) a[8 + i] = p[16 + i]; // K = kh + 16..23
  return a;
}

// Cooperative copy of a [128K x 128N] fp32 weight block (row-major, leading
// dim ldn) into LDS as f16, transposed to [n][k] (ld = 136 halves).
// Each thread handles K row-pairs so the transposed scatter becomes packed
// v_cvt_pk_f16_f32 + aligned ds_store_b32 (32 stores/thread instead of 64).
__device__ inline void stageW(const float* __restrict__ W, int ldn, int k0,
                              int n0, _Float16 (*Ws)[136], int tid) {
#pragma unroll
  for (int p = 0; p < 8; ++p) {
    const int kp = p * 16 + ((tid >> 5) << 1);      // even k in 0..126
    const int n4 = (tid & 31) * 4;
    const float4 a = *(const float4*)(W + (size_t)(k0 + kp) * ldn + n0 + n4);
    const float4 b =
        *(const float4*)(W + (size_t)(k0 + kp + 1) * ldn + n0 + n4);
    union { _Float16 h[2]; unsigned u; } t0, t1, t2, t3;
    t0.h[0] = (_Float16)a.x; t0.h[1] = (_Float16)b.x;
    t1.h[0] = (_Float16)a.y; t1.h[1] = (_Float16)b.y;
    t2.h[0] = (_Float16)a.z; t2.h[1] = (_Float16)b.z;
    t3.h[0] = (_Float16)a.w; t3.h[1] = (_Float16)b.w;
    *(unsigned*)&Ws[n4 + 0][kp] = t0.u;
    *(unsigned*)&Ws[n4 + 1][kp] = t1.u;
    *(unsigned*)&Ws[n4 + 2][kp] = t2.u;
    *(unsigned*)&Ws[n4 + 3][kp] = t3.u;
  }
}

// ---------------------------------------------------------------------------
// Kernel 1: zero output (we later do exactly two commutative atomic adds).
// ---------------------------------------------------------------------------
__global__ void zero_f32(float* __restrict__ p, size_t n) {
  size_t i = (size_t)blockIdx.x * blockDim.x + threadIdx.x;
  const size_t stride = (size_t)gridDim.x * blockDim.x;
  for (; i < n; i += stride) p[i] = 0.0f;
}

// ---------------------------------------------------------------------------
// Kernel 2: router. One wave32 per token, shuffle-tree reduce, lane 0 does
// top-2 (first-index tie-break, like jax.lax.top_k) + 2-way softmax.
// ---------------------------------------------------------------------------
__global__ __launch_bounds__(256) void router_topk(
    const float* __restrict__ X, const float* __restrict__ Wg,
    int* __restrict__ topk_idx, float* __restrict__ topk_wt) {
  constexpr int C = 1024, E = 8;
  const int lane  = threadIdx.x & 31;
  const int token = blockIdx.x * 8 + (threadIdx.x >> 5);
  const float* xrow = X + (size_t)token * C;

  float acc[E];
#pragma unroll
  for (int e = 0; e < E; ++e) acc[e] = 0.0f;

  for (int c = lane; c < C; c += 32) {
    const float xv = xrow[c];
#pragma unroll
    for (int e = 0; e < E; ++e) acc[e] += xv * Wg[c * E + e];
  }
#pragma unroll
  for (int e = 0; e < E; ++e) {
#pragma unroll
    for (int m = 16; m >= 1; m >>= 1) acc[e] += __shfl_xor(acc[e], m, 32);
  }
  if (lane == 0) {
    int i0 = 0; float v0 = acc[0];
    for (int e = 1; e < E; ++e) if (acc[e] > v0) { v0 = acc[e]; i0 = e; }
    int i1 = -1; float v1 = -3.4e38f;
    for (int e = 0; e < E; ++e)
      if (e != i0 && acc[e] > v1) { v1 = acc[e]; i1 = e; }
    const float g1 = 1.0f / (1.0f + expf(v0 - v1));
    topk_idx[2 * token]     = i0;
    topk_idx[2 * token + 1] = i1;
    topk_wt[2 * token]      = 1.0f - g1;
    topk_wt[2 * token + 1]  = g1;
  }
}

// ---------------------------------------------------------------------------
// Kernel 3: deterministic per-expert compaction via block-wide scan.
// ---------------------------------------------------------------------------
__global__ __launch_bounds__(256) void bucketize(
    const int* __restrict__ topk_idx, const float* __restrict__ topk_wt,
    int* __restrict__ bucket_tok, float* __restrict__ bucket_wt,
    int* __restrict__ counts) {
  constexpr int NTOK = 16384;
  const int e   = blockIdx.x;
  const int tid = threadIdx.x;
  __shared__ int scan[256];

  int running = 0;
  for (int base = 0; base < NTOK; base += 256) {
    const int t  = base + tid;
    const int i0 = topk_idx[2 * t];
    const int i1 = topk_idx[2 * t + 1];
    const int pred = (i0 == e) || (i1 == e);
    const float w  = (i0 == e) ? topk_wt[2 * t] : topk_wt[2 * t + 1];
    scan[tid] = pred;
    __syncthreads();
#pragma unroll
    for (int off = 1; off < 256; off <<= 1) {
      const int v = (tid >= off) ? scan[tid - off] : 0;
      __syncthreads();
      scan[tid] += v;
      __syncthreads();
    }
    if (pred) {
      const int pos = running + scan[tid] - 1;
      bucket_tok[(size_t)e * NTOK + pos] = t;
      bucket_wt[(size_t)e * NTOK + pos]  = w;
    }
    running += scan[255];
    __syncthreads();
  }
  if (tid == 0) counts[e] = running;
}

// ---------------------------------------------------------------------------
// Kernel 4: fused expert FFN, WMMA f16, double-buffered LDS-staged weights.
// Block = 32 routed tokens (2 M-tiles) of expert e, 8 waves.
// Per 128-wide D chunk:
//   phase A (8 K-subblocks of C): stage(next Ws buf) -> wmma(cur buf) -> one
//           barrier.  h += Xs @ W1; then bias + exact GELU -> Hs (32x128 f16).
//   phase C (8 N-subblocks of C): same double-buffered flow; wave w owns
//           column tile nsub*8+w: acc[nsub][mt] += Hs @ W2.
// Epilogue: out[tok,col] += gate * (acc + b2[col]) via atomicAdd (exactly two
// commutative f32 adds per element across the dispatch -> bit-deterministic).
// LDS: Xs 66.0K + Hs 8.7K + Ws 69.6K  ~= 144.4 KB (<320 KB/WGP).
// ---------------------------------------------------------------------------
__global__ __launch_bounds__(256) void expert_ffn(
    const float* __restrict__ X,  const float* __restrict__ W1,
    const float* __restrict__ b1, const float* __restrict__ W2,
    const float* __restrict__ b2, const int* __restrict__ bucket_tok,
    const float* __restrict__ bucket_wt, const int* __restrict__ counts,
    float* __restrict__ out) {
  constexpr int C = 1024, D = 4096, NTOK = 16384;
  const int e     = blockIdx.y;
  const int cnt   = counts[e];
  const int start = blockIdx.x * 32;
  if (start >= cnt) return;  // uniform exit: EXEC all-1s at every WMMA

  __shared__ __align__(16) _Float16 Xs[32][1032];     // 66.0 KB
  __shared__ __align__(16) _Float16 Hs[32][136];      //  8.7 KB
  __shared__ __align__(16) _Float16 Ws[2][128][136];  // 69.6 KB, dbl-buffered
  __shared__ int   toks[32];
  __shared__ float wts[32];

  const int tid  = threadIdx.x;
  const int lane = tid & 31;
  const int wv   = tid >> 5;

  if (tid < 32) {
    const int idx = start + tid;
    toks[tid] = (idx < cnt) ? bucket_tok[(size_t)e * NTOK + idx] : -1;
    wts[tid]  = (idx < cnt) ? bucket_wt[(size_t)e * NTOK + idx] : 0.0f;
  }
  __syncthreads();

  for (int m = 0; m < 32; ++m) {
    const int tok = toks[m];
    const float* xrow = X + (size_t)(tok < 0 ? 0 : tok) * C;
    for (int c = tid; c < C; c += 256)
      Xs[m][c] = (tok >= 0) ? (_Float16)xrow[c] : (_Float16)0.0f;
  }

  const float* w1e = W1 + (size_t)e * C * D;
  const float* w2e = W2 + (size_t)e * D * C;
  const float* b1e = b1 + (size_t)e * D;
  const float* b2e = b2 + (size_t)e * C;

  v8f acc2[8][2];
#pragma unroll
  for (int i = 0; i < 8; ++i)
#pragma unroll
    for (int mt = 0; mt < 2; ++mt)
#pragma unroll
      for (int j = 0; j < 8; ++j) acc2[i][mt][j] = 0.0f;

#pragma unroll 1
  for (int ch = 0; ch < D / 128; ++ch) {
    const int d0 = ch * 128;
    __builtin_prefetch(w1e + (size_t)lane * D + d0, 0, 1);

    // ---- phase A: H(32x128) = gelu(Xs(32x1024) @ W1[:, d0:d0+128] + b1) --
    v8f h0, h1;
#pragma unroll
    for (int j = 0; j < 8; ++j) { h0[j] = 0.0f; h1[j] = 0.0f; }

    stageW(w1e, D, 0, d0, Ws[0], tid);    // prologue (also covers Xs writes)
    __syncthreads();
#pragma unroll 1
    for (int ksub = 0; ksub < C / 128; ++ksub) {
      if (ksub < 7)
        stageW(w1e, D, (ksub + 1) * 128, d0, Ws[(ksub + 1) & 1], tid);
#pragma unroll
      for (int kk = 0; kk < 4; ++kk) {
        const v16h a0 = ldsA(&Xs[0][0],  1032, ksub * 128 + kk * 32, lane);
        const v16h a1 = ldsA(&Xs[16][0], 1032, ksub * 128 + kk * 32, lane);
        const v16h b  = ldsA(&Ws[ksub & 1][wv * 16][0], 136, kk * 32, lane);
        h0 = __builtin_amdgcn_wmma_f32_16x16x32_f16(false, a0, false, b,
                                                    (short)0, h0, false, false);
        h1 = __builtin_amdgcn_wmma_f32_16x16x32_f16(false, a1, false, b,
                                                    (short)0, h1, false, false);
      }
      __syncthreads();  // next buf staged by all; cur buf fully consumed
    }

    // phase C prologue staged concurrently with the GELU epilogue of phase A
    stageW(w2e, C, d0, 0, Ws[0], tid);
    const float b1v = b1e[d0 + wv * 16 + (lane & 15)];
#pragma unroll
    for (int r = 0; r < 8; ++r) {
      const int mlo = r + ((lane >> 4) << 3);
      float v = h0[r] + b1v;
      v = 0.5f * v * (1.0f + erff(v * 0.70710678118654752f));
      Hs[mlo][wv * 16 + (lane & 15)] = (_Float16)v;
      float u = h1[r] + b1v;
      u = 0.5f * u * (1.0f + erff(u * 0.70710678118654752f));
      Hs[16 + mlo][wv * 16 + (lane & 15)] = (_Float16)u;
    }
    __syncthreads();  // Hs ready AND Ws[0] staged

    // Hoist H fragments (reused across all 8 N-subblocks).
    v16h aH[4][2];
#pragma unroll
    for (int kk = 0; kk < 4; ++kk) {
      aH[kk][0] = ldsA(&Hs[0][0],  136, kk * 32, lane);
      aH[kk][1] = ldsA(&Hs[16][0], 136, kk * 32, lane);
    }

    // ---- phase C: acc[nsub][mt] += Hs @ W2[d0:d0+128, nsub*128 + wv*16] --
#pragma unroll 1
    for (int nsub = 0; nsub < 8; ++nsub) {
      if (nsub < 7)
        stageW(w2e, C, d0, (nsub + 1) * 128, Ws[(nsub + 1) & 1], tid);
#pragma unroll
      for (int kk = 0; kk < 4; ++kk) {
        const v16h b = ldsA(&Ws[nsub & 1][wv * 16][0], 136, kk * 32, lane);
        acc2[nsub][0] = __builtin_amdgcn_wmma_f32_16x16x32_f16(
            false, aH[kk][0], false, b, (short)0, acc2[nsub][0], false, false);
        acc2[nsub][1] = __builtin_amdgcn_wmma_f32_16x16x32_f16(
            false, aH[kk][1], false, b, (short)0, acc2[nsub][1], false, false);
      }
      __syncthreads();
    }
  }

  // ---- epilogue: wave w owns column tiles nsub*8 + w ----------------------
#pragma unroll
  for (int nsub = 0; nsub < 8; ++nsub) {
    const int col = (nsub * 8 + wv) * 16 + (lane & 15);
    const float b2v = b2e[col];
#pragma unroll
    for (int mt = 0; mt < 2; ++mt) {
#pragma unroll
      for (int r = 0; r < 8; ++r) {
        const int m   = mt * 16 + r + ((lane >> 4) << 3);
        const int tok = toks[m];
        if (tok >= 0)
          atomicAdd(&out[(size_t)tok * C + col],
                    wts[m] * (acc2[nsub][mt][r] + b2v));
      }
    }
  }
}

// ---------------------------------------------------------------------------
extern "C" void kernel_launch(void* const* d_in, const int* in_sizes, int n_in,
                              void* d_out, int out_size, void* d_ws,
                              size_t ws_size, hipStream_t stream) {
  constexpr int NTOK = 16384, C = 1024, E = 8;
  const float* X  = (const float*)d_in[0];
  const float* Wg = (const float*)d_in[1];
  const float* W1 = (const float*)d_in[2];
  const float* b1 = (const float*)d_in[3];
  const float* W2 = (const float*)d_in[4];
  const float* b2 = (const float*)d_in[5];
  float* out = (float*)d_out;

  char* p = (char*)d_ws;
  float* topk_wt   = (float*)p; p += (size_t)NTOK * 2 * sizeof(float);
  int*   topk_idx  = (int*)p;   p += (size_t)NTOK * 2 * sizeof(int);
  int*   counts    = (int*)p;   p += 256;
  int*   bucket_tok = (int*)p;  p += (size_t)E * NTOK * sizeof(int);
  float* bucket_wt  = (float*)p;

  zero_f32<<<4096, 256, 0, stream>>>(out, (size_t)NTOK * C);
  router_topk<<<NTOK / 8, 256, 0, stream>>>(X, Wg, topk_idx, topk_wt);
  bucketize<<<E, 256, 0, stream>>>(topk_idx, topk_wt, bucket_tok, bucket_wt,
                                   counts);
  expert_ffn<<<dim3(NTOK / 32, E), 256, 0, stream>>>(
      X, W1, b1, W2, b2, bucket_tok, bucket_wt, counts, out);
}
